// Qwen3AttentionSeparated_76725295775935
// MI455X (gfx1250) — matrix operand order repaired and checked by reading, not compiled, not run on hardware
//
#include <hip/hip_runtime.h>
#include <math.h>

#define S_LEN   8192
#define HID     4096
#define NH      32
#define NKV     8
#define HD      128
#define QKV_N   6144          // (32 + 2*8) * 128
#define SCALE_F 0.08838834764831845f   // 128^-0.5
#define EPS_F   1e-6f

typedef unsigned short u16;
typedef unsigned int   u32;
typedef __attribute__((ext_vector_type(16))) __bf16 bf16x16;
typedef __attribute__((ext_vector_type(8)))  float  floatx8;
typedef __attribute__((ext_vector_type(4)))  u32    u32x4;
typedef int v4i __attribute__((vector_size(16)));   // builtin's pointee type

struct B32Frag { u32x4 lo, hi; };   // 32 bytes == one bf16x16 operand

#define AS1 __attribute__((address_space(1)))
#define AS3 __attribute__((address_space(3)))

#if defined(__has_builtin)
#if __has_builtin(__builtin_amdgcn_global_load_async_to_lds_b128) && \
    __has_builtin(__builtin_amdgcn_s_wait_asynccnt)
#define HAVE_ASYNC_LDS 1
#endif
#endif
#ifndef HAVE_ASYNC_LDS
#define HAVE_ASYNC_LDS 0
#endif

__device__ __forceinline__ u16 f2bf(float f) {
  u32 u = __float_as_uint(f);
  u32 r = u + 0x7FFFu + ((u >> 16) & 1u);   // round-to-nearest-even
  return (u16)(r >> 16);
}
__device__ __forceinline__ float b2f(u16 h) {
  return __uint_as_float(((u32)h) << 16);
}

// 16-byte global -> LDS async copy (gfx1250 GLOBAL_LOAD_ASYNC_TO_LDS_B128,
// tracked by ASYNCcnt).  No VGPR round trip.
__device__ __forceinline__ void async_cp16(const u16* g, u16* l) {
#if HAVE_ASYNC_LDS
  __builtin_amdgcn_global_load_async_to_lds_b128(
      (AS1 v4i*)g, (AS3 v4i*)l, /*imm offset*/ 0, /*cpol*/ 0);
#else
  *(u32x4*)l = *(const u32x4*)g;   // sync fallback (reg staging)
#endif
}
template <int N> __device__ __forceinline__ void wait_async() {
#if HAVE_ASYNC_LDS
  __builtin_amdgcn_s_wait_asynccnt((unsigned short)N);
#endif
}

// ---------------------------------------------------------------------------
// fp32 -> bf16 conversion, 4 elements/thread (b128 loads, b64 stores)
// ---------------------------------------------------------------------------
__global__ __launch_bounds__(256) void cvt_f32_bf16(const float4* __restrict__ in,
                                                    u32* __restrict__ out, int n4) {
  int i = blockIdx.x * blockDim.x + threadIdx.x;
  const int stride = gridDim.x * blockDim.x;
  for (; i < n4; i += stride) {
    const float4 v = in[i];
    out[2 * i + 0] = (u32)f2bf(v.x) | ((u32)f2bf(v.y) << 16);
    out[2 * i + 1] = (u32)f2bf(v.z) | ((u32)f2bf(v.w) << 16);
  }
}

// ---------------------------------------------------------------------------
// RMSNorm over hidden (4096) + cast to bf16.  One block per token.
// ---------------------------------------------------------------------------
__global__ __launch_bounds__(256) void rmsnorm_hidden(const float* __restrict__ x,
                                                      const float* __restrict__ w,
                                                      u16* __restrict__ y) {
  const int s = blockIdx.x;
  const float* row = x + (size_t)s * HID;
  float v[16];
  float ss = 0.0f;
#pragma unroll
  for (int r = 0; r < 16; ++r) {
    v[r] = row[threadIdx.x + 256 * r];
    ss += v[r] * v[r];
  }
  for (int off = 16; off; off >>= 1) ss += __shfl_xor(ss, off);
  __shared__ float red[8];
  if ((threadIdx.x & 31) == 0) red[threadIdx.x >> 5] = ss;
  __syncthreads();
  float tot = 0.0f;
#pragma unroll
  for (int i = 0; i < 8; ++i) tot += red[i];
  const float inv = rsqrtf(tot * (1.0f / HID) + EPS_F);
  u16* yrow = y + (size_t)s * HID;
#pragma unroll
  for (int r = 0; r < 16; ++r) {
    const int i = threadIdx.x + 256 * r;
    yrow[i] = f2bf(v[r] * inv * w[i]);
  }
}

// ---------------------------------------------------------------------------
// bf16 WMMA GEMM:  C[M,N] = A[M,K] * B[N,K]^T (+ bias)
// Block tile 128x128, BK=32, 8 wave32 waves, each owning 64x32 of C
// (4x2 tiles of v_wmma_f32_16x16x32_bf16).  Double-buffered LDS fed by
// async global->LDS DMA (ASYNCcnt) so tile k+1 streams in under tile k's WMMAs.
// ---------------------------------------------------------------------------
template <bool STORE_BF16>
__global__ __launch_bounds__(256) void gemm_bf16_wmma(
    const u16* __restrict__ A, const u16* __restrict__ B,
    const float* __restrict__ bias,
    float* __restrict__ Cf, u16* __restrict__ Cb,
    int M, int N, int K) {
  constexpr int BM = 128, BN = 128, BK = 32, LK = BK + 8;  // pad -> 20-bank row stride
  __shared__ u16 As[2][BM * LK];
  __shared__ u16 Bs[2][BN * LK];

  const int t    = threadIdx.x;
  const int lane = t & 31;
  const int wid  = t >> 5;
  const int half = lane >> 4;       // 0 | 1
  const int l16  = lane & 15;       // 0..15
  const int wm   = wid & 1;         // 2 waves along M (64 rows each)
  const int wn   = wid >> 1;        // 4 waves along N (32 cols each)
  const int mBase = blockIdx.y * BM;
  const int nBase = blockIdx.x * BN;

  floatx8 acc[4][2];
#pragma unroll
  for (int i = 0; i < 4; ++i)
#pragma unroll
    for (int j = 0; j < 2; ++j)
#pragma unroll
      for (int r = 0; r < 8; ++r) acc[i][j][r] = 0.0f;

  // Cooperative loader: thread t owns 32B of A and 32B of B per K-tile.
  const int ldRow = t >> 1;           // 0..127
  const int ldCol = (t & 1) * 16;     // 0 | 16 elements
  const u16* aG = A + (size_t)(mBase + ldRow) * K + ldCol;
  const u16* bG = B + (size_t)(nBase + ldRow) * K + ldCol;
  const int ldsOff = ldRow * LK + ldCol;

  auto issue_tile = [&](int k0, int buf) {
    async_cp16(aG + k0,     &As[buf][ldsOff]);
    async_cp16(aG + k0 + 8, &As[buf][ldsOff + 8]);
    async_cp16(bG + k0,     &Bs[buf][ldsOff]);
    async_cp16(bG + k0 + 8, &Bs[buf][ldsOff + 8]);
  };

  const int KT = K / BK;
  issue_tile(0, 0);
  for (int kt = 0; kt < KT; ++kt) {
    const int cur = kt & 1;
    if (kt + 1 < KT) {
      issue_tile((kt + 1) * BK, cur ^ 1);   // overlaps with this tile's WMMAs
      wait_async<4>();                      // current tile's 4 ops are the oldest
    } else {
      wait_async<0>();
    }
    if (kt + 2 < KT) {                      // keep tile k+2 warming the caches
      __builtin_prefetch(aG + (kt + 2) * BK, 0, 1);
      __builtin_prefetch(bG + (kt + 2) * BK, 0, 1);
    }
    __syncthreads();                        // tile data visible to all waves

    // A fragment (16x32 MxK): lane l16 = M; e -> K = half*16 + e (contiguous).
    bf16x16 afrag[4];
#pragma unroll
    for (int i = 0; i < 4; ++i) {
      const u16* p = &As[cur][(wm * 64 + i * 16 + l16) * LK + half * 16];
      B32Frag f; f.lo = *(const u32x4*)p; f.hi = *(const u32x4*)(p + 8);
      afrag[i] = __builtin_bit_cast(bf16x16, f);
    }
    // B fragment (32x16 KxN): lane l16 = N; e -> K = half*16 + e (contiguous).
    bf16x16 bfrag[2];
#pragma unroll
    for (int j = 0; j < 2; ++j) {
      const u16* p = &Bs[cur][(wn * 32 + j * 16 + l16) * LK + half * 16];
      B32Frag f; f.lo = *(const u32x4*)p; f.hi = *(const u32x4*)(p + 8);
      bfrag[j] = __builtin_bit_cast(bf16x16, f);
    }
#pragma unroll
    for (int i = 0; i < 4; ++i)
#pragma unroll
      for (int j = 0; j < 2; ++j)
        acc[i][j] = __builtin_amdgcn_wmma_f32_16x16x32_bf16(
            false, afrag[i], false, bfrag[j], (short)0, acc[i][j], false, false);
    __syncthreads();   // all reads done before buf[cur] is re-issued at kt+2
  }

  // C layout: VGPR r, lane -> (M = r + 8*half, N = l16)
#pragma unroll
  for (int i = 0; i < 4; ++i) {
#pragma unroll
    for (int j = 0; j < 2; ++j) {
      const int n  = nBase + wn * 32 + j * 16 + l16;
      const float bv = bias ? bias[n] : 0.0f;
      const int m0 = mBase + wm * 64 + i * 16 + half * 8;
#pragma unroll
      for (int r = 0; r < 8; ++r) {
        const float v = acc[i][j][r] + bv;
        const size_t idx = (size_t)(m0 + r) * N + n;
        if constexpr (STORE_BF16) Cb[idx] = f2bf(v);
        else                      Cf[idx] = v;
      }
    }
  }
}

// ---------------------------------------------------------------------------
// RoPE + per-head RMSNorm for q (32 heads) and k (8 heads), in-place on bf16
// qkv buffer.  One block (256 thr) per token; 2 heads per iteration.
// ---------------------------------------------------------------------------
__global__ __launch_bounds__(256) void rope_qknorm(
    u16* __restrict__ qkv, const int* __restrict__ pos,
    const float* __restrict__ cosc, const float* __restrict__ sinc,
    const float* __restrict__ qw, const float* __restrict__ kw) {
  const int s = blockIdx.x;
  const int p = pos[s];
  u16* row = qkv + (size_t)s * QKV_N;
  const int t = threadIdx.x;
  const int d = t & 127;
  const int hslot = t >> 7;            // which of 2 heads this iteration
  const int j = d & 63;
  const float c  = cosc[(size_t)p * 64 + j];
  const float sn = sinc[(size_t)p * 64 + j];
  __shared__ float red[2][4];

  for (int it = 0; it < 20; ++it) {    // 40 rope'd heads total
    const int hh = it * 2 + hslot;
    const int base = (hh < NH) ? hh * HD : HID + (hh - NH) * HD;
    const float* nw = (hh < NH) ? qw : kw;
    const float x  = b2f(row[base + d]);
    const float xp = b2f(row[base + ((d < 64) ? d + 64 : d - 64)]);
    const float r  = (d < 64) ? (x * c - xp * sn) : (x * c + xp * sn);
    float ss = r * r;
    for (int off = 16; off; off >>= 1) ss += __shfl_xor(ss, off);
    if ((t & 31) == 0) red[hslot][(t >> 5) & 3] = ss;
    __syncthreads();                   // also orders rope reads before writes
    const float tot = red[hslot][0] + red[hslot][1] + red[hslot][2] + red[hslot][3];
    const float inv = rsqrtf(tot * (1.0f / HD) + EPS_F);
    row[base + d] = f2bf(r * inv * nw[d]);
    __syncthreads();
  }
}

// ---------------------------------------------------------------------------
// Per-token head attention: scores[h][g] = q[h].k[g>>2] * SCALE, softmax over
// g, ctx[h] = sum_g p * v[g>>2].  One block per token (tiny: 0.6% of FLOPs).
// ---------------------------------------------------------------------------
__global__ __launch_bounds__(256) void head_attn(const u16* __restrict__ qkv,
                                                 u16* __restrict__ ctx) {
  const int s = blockIdx.x;
  const u16* row = qkv + (size_t)s * QKV_N;
  __shared__ float qS[NH * HD];        // 16 KB
  __shared__ float kS[NKV * HD];       // 4 KB
  __shared__ float vS[NKV * HD];       // 4 KB
  __shared__ float pS[NH * NH];        // 4 KB
  const int t = threadIdx.x;
  for (int i = t; i < NH * HD; i += 256) qS[i] = b2f(row[i]);
  for (int i = t; i < NKV * HD; i += 256) {
    kS[i] = b2f(row[HID + i]);
    vS[i] = b2f(row[(NH + NKV) * HD + i]);
  }
  __syncthreads();

  const int h = t >> 3, sub = t & 7;   // 8 lanes cooperate per query head
  float sc[4];
#pragma unroll
  for (int gi = 0; gi < 4; ++gi) {
    const int g = sub + gi * 8;
    const int kv = g >> 2;             // GQA: repeat(k, 4)
    float acc = 0.0f;
    for (int dd = 0; dd < HD; ++dd) acc += qS[h * HD + dd] * kS[kv * HD + dd];
    sc[gi] = acc * SCALE_F;
  }
  float mx = fmaxf(fmaxf(sc[0], sc[1]), fmaxf(sc[2], sc[3]));
  for (int off = 1; off < 8; off <<= 1) mx = fmaxf(mx, __shfl_xor(mx, off));
  float e[4], sum = 0.0f;
#pragma unroll
  for (int gi = 0; gi < 4; ++gi) { e[gi] = expf(sc[gi] - mx); sum += e[gi]; }
  for (int off = 1; off < 8; off <<= 1) sum += __shfl_xor(sum, off);
  const float rs = 1.0f / sum;
#pragma unroll
  for (int gi = 0; gi < 4; ++gi) pS[h * NH + sub + gi * 8] = e[gi] * rs;
  __syncthreads();

  u16* crow = ctx + (size_t)s * (NH * HD);
  const int dbase = sub * 16;
#pragma unroll
  for (int dd = 0; dd < 16; ++dd) {
    const int d = dbase + dd;
    float acc = 0.0f;
#pragma unroll
    for (int g = 0; g < NH; ++g) acc += pS[h * NH + g] * vS[(g >> 2) * HD + d];
    crow[h * HD + d] = f2bf(acc);
  }
}

// ---------------------------------------------------------------------------
extern "C" void kernel_launch(void* const* d_in, const int* in_sizes, int n_in,
                              void* d_out, int out_size, void* d_ws, size_t ws_size,
                              hipStream_t stream) {
  (void)in_sizes; (void)n_in; (void)out_size; (void)ws_size;
  const int*   positions = (const int*)d_in[0];
  const float* hidden    = (const float*)d_in[1];
  const float* lw        = (const float*)d_in[2];
  const float* wqkv      = (const float*)d_in[3];
  const float* bqkv      = (const float*)d_in[4];
  const float* wo        = (const float*)d_in[5];
  const float* qnw       = (const float*)d_in[6];
  const float* knw       = (const float*)d_in[7];
  const float* cosc      = (const float*)d_in[8];
  const float* sinc      = (const float*)d_in[9];
  float* out = (float*)d_out;

  size_t off = 0;
  auto walloc = [&](size_t bytes) -> void* {
    void* p = (char*)d_ws + off;
    off += (bytes + 255) & ~(size_t)255;
    return p;
  };
  u16* xnorm = (u16*)walloc((size_t)S_LEN * HID   * 2);  //  64 MB
  u16* wqkvb = (u16*)walloc((size_t)QKV_N * HID   * 2);  //  48 MB
  u16* wob   = (u16*)walloc((size_t)HID   * HID   * 2);  //  32 MB
  u16* qkvb  = (u16*)walloc((size_t)S_LEN * QKV_N * 2);  //  96 MB
  u16* ctxb  = (u16*)walloc((size_t)S_LEN * HID   * 2);  //  64 MB

  cvt_f32_bf16<<<4096, 256, 0, stream>>>((const float4*)wqkv, (u32*)wqkvb,
                                         QKV_N * HID / 4);
  cvt_f32_bf16<<<4096, 256, 0, stream>>>((const float4*)wo, (u32*)wob,
                                         HID * HID / 4);
  rmsnorm_hidden<<<S_LEN, 256, 0, stream>>>(hidden, lw, xnorm);
  gemm_bf16_wmma<true><<<dim3(QKV_N / 128, S_LEN / 128), 256, 0, stream>>>(
      xnorm, wqkvb, bqkv, nullptr, qkvb, S_LEN, QKV_N, HID);
  rope_qknorm<<<S_LEN, 256, 0, stream>>>(qkvb, positions, cosc, sinc, qnw, knw);
  head_attn<<<S_LEN, 256, 0, stream>>>(qkvb, ctxb);
  gemm_bf16_wmma<false><<<dim3(HID / 128, S_LEN / 128), 256, 0, stream>>>(
      ctxb, wob, nullptr, out, nullptr, S_LEN, HID, HID);
}
